// DocumentHead_83528523973131
// MI455X (gfx1250) — matrix operand order repaired and checked by reading, not compiled, not run on hardware
//
#include <hip/hip_runtime.h>
#include <hip/hip_bf16.h>

typedef __attribute__((ext_vector_type(16))) __bf16 v16bf;
typedef __attribute__((ext_vector_type(8)))  __bf16 v8bf;
typedef __attribute__((ext_vector_type(4)))  __bf16 v4bf;
typedef __attribute__((ext_vector_type(8)))  float  v8f;
typedef __attribute__((ext_vector_type(4)))  unsigned int u32x4;

#define DIM 2048
#define EPS 1e-12f

#if defined(__HIP_DEVICE_COMPILE__) && __has_builtin(__builtin_amdgcn_global_load_async_to_lds_b128)
#define HAVE_ASYNC_LDS 1
#else
#define HAVE_ASYNC_LDS 0
#endif

#if HAVE_ASYNC_LDS
// per hipcc diagnostic: arg0 is 'int __attribute__((vector_size(16))) __device__ *'
typedef __attribute__((__vector_size__(16))) int i32x4v;
typedef __attribute__((address_space(1))) i32x4v* g_v4i_ptr;
typedef __attribute__((address_space(3))) i32x4v* l_v4i_ptr;
typedef const __attribute__((address_space(1))) void* g_cvoid_ptr;
typedef __attribute__((address_space(3))) void* l_void_ptr;

__device__ __forceinline__ void async_copy16(const void* g, void* l) {
  __builtin_amdgcn_global_load_async_to_lds_b128(
      (g_v4i_ptr)(g_cvoid_ptr)g,   // generic->AS(1), then to int4* (drops const)
      (l_v4i_ptr)(l_void_ptr)l,    // generic->AS(3), then to int4*
      0, 0);
}

#if __has_builtin(__builtin_amdgcn_s_wait_asynccnt)
#define WAIT_ASYNC0() __builtin_amdgcn_s_wait_asynccnt(0)
#else
#define WAIT_ASYNC0() asm volatile("s_wait_asynccnt 0" ::: "memory")
#endif
#endif // HAVE_ASYNC_LDS

// ---------------------------------------------------------------------------
// Kernel 1: per-row cosine stats: align, div=1-align, tens=div^2
// sc layout: [0..B) align, [B..2B) div, [2B..3B) tens
// ---------------------------------------------------------------------------
__global__ void row_scalars_kernel(const float* __restrict__ vc,
                                   const float* __restrict__ vd,
                                   float* __restrict__ sc, int B) {
  __shared__ float rcc[256], rdd[256], rcd[256];
  const int r = blockIdx.x;
  const float* c = vc + (size_t)r * DIM;
  const float* d = vd + (size_t)r * DIM;
  float cc = 0.f, dd = 0.f, cd = 0.f;
  for (int i = threadIdx.x; i < DIM; i += 256) {
    float x = c[i], y = d[i];
    cc += x * x; dd += y * y; cd += x * y;
  }
  rcc[threadIdx.x] = cc; rdd[threadIdx.x] = dd; rcd[threadIdx.x] = cd;
  __syncthreads();
  for (int s = 128; s > 0; s >>= 1) {
    if (threadIdx.x < s) {
      rcc[threadIdx.x] += rcc[threadIdx.x + s];
      rdd[threadIdx.x] += rdd[threadIdx.x + s];
      rcd[threadIdx.x] += rcd[threadIdx.x + s];
    }
    __syncthreads();
  }
  if (threadIdx.x == 0) {
    float nc = fmaxf(sqrtf(rcc[0]), EPS);
    float nd = fmaxf(sqrtf(rdd[0]), EPS);
    float align = rcd[0] / (nc * nd);
    float div = 1.0f - align;
    sc[r] = align; sc[B + r] = div; sc[2 * B + r] = div * div;
  }
}

// ---------------------------------------------------------------------------
// Kernel 2: fp32 -> bf16 convert, 4-wide vectorized
// ---------------------------------------------------------------------------
__global__ void cvt_bf16_kernel(const float* __restrict__ src,
                                __bf16* __restrict__ dst, long n4) {
  long i = (long)blockIdx.x * blockDim.x + threadIdx.x;
  const long stride = (long)gridDim.x * blockDim.x;
  for (; i < n4; i += stride) {
    float4 f = ((const float4*)src)[i];
    v4bf o;
    o.x = (__bf16)f.x; o.y = (__bf16)f.y; o.z = (__bf16)f.z; o.w = (__bf16)f.w;
    ((v4bf*)dst)[i] = o;
  }
}

// ---------------------------------------------------------------------------
// Kernel 3: fp32 [K][N] -> bf16 transposed [N][K]
// ---------------------------------------------------------------------------
__global__ void transpose_cvt_kernel(const float* __restrict__ src,
                                     __bf16* __restrict__ dst, int K, int N) {
  const long total = (long)K * N;
  for (long o = (long)blockIdx.x * blockDim.x + threadIdx.x; o < total;
       o += (long)gridDim.x * blockDim.x) {
    long n = o / K, k = o % K;
    dst[o] = (__bf16)src[k * (long)N + n];
  }
}

// ---------------------------------------------------------------------------
// WMMA fragment loads from LDS (16-byte ds_load_b128 pairs)
// A (16x32 bf16): lane half h, row m: K runs {8h..8h+7, 16+8h..16+8h+7}
// B (32x16 bf16): lane half h, col n: K runs {8h..8h+7, 16+8h..16+8h+7}
// ---------------------------------------------------------------------------
__device__ inline v16bf ld_frag_a(const __bf16* p) {
  v8bf lo = *(const v8bf*)p;
  v8bf hi = *(const v8bf*)(p + 16);
  return __builtin_shufflevector(lo, hi, 0,1,2,3,4,5,6,7,8,9,10,11,12,13,14,15);
}
__device__ inline v16bf ld_frag_b(const __bf16* p) {
  v8bf lo = *(const v8bf*)p;
  v8bf hi = *(const v8bf*)(p + 16);
  return __builtin_shufflevector(lo, hi, 0,1,2,3,4,5,6,7,8,9,10,11,12,13,14,15);
}

// ---------------------------------------------------------------------------
// Kernel 4: bf16 WMMA GEMM, C = relu(A @ Bt^T + init), bf16 out
//   A  : [M][K] bf16 row-major;  Bt : [N][K] bf16;  C : [M][N] bf16
// init = bias[n] (+ align[m]*wex[0][n] + div[m]*wex[1][n] + tens[m]*wex[2][n])
// Block tile 256x128, 8 waves (4M x 2N), wave tile 64x64 (4x4 WMMA tiles)
// => 16 WMMA per 8 fragment loads per K-step.
// Global->LDS via async-LDS loads (ASYNCcnt) when available, double-buffered.
// ---------------------------------------------------------------------------
__global__ __launch_bounds__(256)
void gemm_bf16_wmma(const __bf16* __restrict__ A,
                    const __bf16* __restrict__ Bt,
                    __bf16* __restrict__ C,
                    int M, int N, int K,
                    const float* __restrict__ bias,
                    const float* __restrict__ sc, int scStride,
                    const float* __restrict__ wex) {
  __shared__ __bf16 sA[2][256 * 40];   // 256 rows x 32 (pad to 40)
  __shared__ __bf16 sB[2][128 * 40];   // 128 rows x 32 (pad to 40)

  const int tid  = threadIdx.x;
  const int lane = tid & 31;
  const int wave = tid >> 5;
  const int wm = wave >> 1;       // 0..3  (M direction, 64 rows each)
  const int wn = wave & 1;        // 0..1  (N direction, 64 cols each)
  const int half = lane >> 4;
  const int lsub = lane & 15;
  const long blockM = (long)blockIdx.y * 256;
  const long blockN = (long)blockIdx.x * 128;

  // staging: 16B chunks; A: 1024 chunks (4/thread), B: 512 chunks (2/thread)
  const int rw   = tid >> 2;            // 0..63
  const int col8 = (tid & 3) * 8;       // 0,8,16,24 (bf16 elems)
  const __bf16* aP[4];
  const __bf16* bP[2];
  int sOffA[4], sOffB[2];
#pragma unroll
  for (int i = 0; i < 4; ++i) {
    aP[i]    = A + (blockM + rw + 64l * i) * (long)K + col8;
    sOffA[i] = (rw + 64 * i) * 40 + col8;
  }
#pragma unroll
  for (int i = 0; i < 2; ++i) {
    bP[i]    = Bt + (blockN + rw + 64l * i) * (long)K + col8;
    sOffB[i] = (rw + 64 * i) * 40 + col8;
  }

  // ---- accumulator init: bias + rank-3 feature correction ----
  float bn[4], wa[4], wd[4], wt[4];
#pragma unroll
  for (int tn = 0; tn < 4; ++tn) {
    int n = (int)blockN + wn * 64 + tn * 16 + lsub;
    bn[tn] = bias[n];
    wa[tn] = wd[tn] = wt[tn] = 0.f;
    if (wex) { wa[tn] = wex[n]; wd[tn] = wex[(long)N + n]; wt[tn] = wex[2l * N + n]; }
  }
  v8f acc[4][4];
#pragma unroll
  for (int tm = 0; tm < 4; ++tm)
#pragma unroll
    for (int v = 0; v < 8; ++v) {
      float a_ = 0.f, d_ = 0.f, t_ = 0.f;
      if (sc) {
        int m = (int)blockM + wm * 64 + tm * 16 + half * 8 + v;
        a_ = sc[m]; d_ = sc[scStride + m]; t_ = sc[2 * scStride + m];
      }
#pragma unroll
      for (int tn = 0; tn < 4; ++tn)
        acc[tm][tn][v] = bn[tn] + a_ * wa[tn] + d_ * wd[tn] + t_ * wt[tn];
    }

  const int NK = K >> 5;

#if HAVE_ASYNC_LDS
  // ---- async global->LDS double-buffered pipeline ----
  auto issue_tile = [&](int buf, int kt) {
    const long koff = (long)kt * 32;
#pragma unroll
    for (int i = 0; i < 4; ++i)
      async_copy16(aP[i] + koff, &sA[buf][sOffA[i]]);
#pragma unroll
    for (int i = 0; i < 2; ++i)
      async_copy16(bP[i] + koff, &sB[buf][sOffB[i]]);
  };
  issue_tile(0, 0);
#else
  u32x4 ra[4], rb[2];
#pragma unroll
  for (int i = 0; i < 4; ++i) ra[i] = *(const u32x4*)aP[i];
#pragma unroll
  for (int i = 0; i < 2; ++i) rb[i] = *(const u32x4*)bP[i];
#pragma unroll
  for (int i = 0; i < 4; ++i) *(u32x4*)&sA[0][sOffA[i]] = ra[i];
#pragma unroll
  for (int i = 0; i < 2; ++i) *(u32x4*)&sB[0][sOffB[i]] = rb[i];
#endif

  for (int kt = 0; kt < NK; ++kt) {
    const int cur = kt & 1;
    const bool more = (kt + 1) < NK;

#if HAVE_ASYNC_LDS
    WAIT_ASYNC0();              // tile `cur` (issued last iteration) landed
    __syncthreads();            // ...for every wave; old reads of nxt done
    if (more) issue_tile(cur ^ 1, kt + 1);  // overlaps with WMMAs below
#else
    if (more) {
      const long koff = (long)(kt + 1) * 32;
#pragma unroll
      for (int i = 0; i < 4; ++i) ra[i] = *(const u32x4*)(aP[i] + koff);
#pragma unroll
      for (int i = 0; i < 2; ++i) rb[i] = *(const u32x4*)(bP[i] + koff);
    }
    __syncthreads();
#endif

    v16bf af[4], bfr[4];
#pragma unroll
    for (int tm = 0; tm < 4; ++tm)
      af[tm] = ld_frag_a(&sA[cur][(wm * 64 + tm * 16 + lsub) * 40 + half * 8]);
#pragma unroll
    for (int tn = 0; tn < 4; ++tn)
      bfr[tn] = ld_frag_b(&sB[cur][(wn * 64 + tn * 16 + lsub) * 40 + half * 8]);

#pragma unroll
    for (int tm = 0; tm < 4; ++tm)
#pragma unroll
      for (int tn = 0; tn < 4; ++tn)
        acc[tm][tn] = __builtin_amdgcn_wmma_f32_16x16x32_bf16(
            false, af[tm], false, bfr[tn], (short)0, acc[tm][tn], false, false);

#if !HAVE_ASYNC_LDS
    if (more) {
      const int nxt = cur ^ 1;
#pragma unroll
      for (int i = 0; i < 4; ++i) *(u32x4*)&sA[nxt][sOffA[i]] = ra[i];
#pragma unroll
      for (int i = 0; i < 2; ++i) *(u32x4*)&sB[nxt][sOffB[i]] = rb[i];
    }
#endif
  }

  // ---- epilogue: ReLU, convert, store ----
#pragma unroll
  for (int tm = 0; tm < 4; ++tm)
#pragma unroll
    for (int tn = 0; tn < 4; ++tn)
#pragma unroll
      for (int v = 0; v < 8; ++v) {
        long m = blockM + wm * 64 + tm * 16 + half * 8 + v;
        long n = blockN + wn * 64 + tn * 16 + lsub;
        float x = acc[tm][tn][v];
        x = x > 0.f ? x : 0.f;
        C[m * (long)N + n] = (__bf16)x;
      }
}

// ---------------------------------------------------------------------------
// Kernel 5: out[m] = sigmoid(dot(S[m,:], Wr) + br) — one wave per row
// ---------------------------------------------------------------------------
__global__ void gemv_sigmoid_kernel(const __bf16* __restrict__ S,
                                    const float* __restrict__ Wr,
                                    const float* __restrict__ br,
                                    float* __restrict__ out, int B, int H) {
  const int wave = threadIdx.x >> 5, lane = threadIdx.x & 31;
  const int r = blockIdx.x * 8 + wave;
  if (r >= B) return;
  const __bf16* s = S + (long)r * H;
  float sum = 0.f;
  for (int k = lane; k < H; k += 32) sum += (float)s[k] * Wr[k];
  for (int off = 16; off > 0; off >>= 1) sum += __shfl_down(sum, off, 32);
  if (lane == 0) out[r] = 1.0f / (1.0f + expf(-(sum + br[0])));
}

// ---------------------------------------------------------------------------
extern "C" void kernel_launch(void* const* d_in, const int* in_sizes, int n_in,
                              void* d_out, int out_size, void* d_ws, size_t ws_size,
                              hipStream_t stream) {
  const float* h_final = (const float*)d_in[0];
  const float* v_claim = (const float*)d_in[1];
  const float* v_doc   = (const float*)d_in[2];
  const float* W1      = (const float*)d_in[3];
  const float* b1      = (const float*)d_in[4];
  const float* W2      = (const float*)d_in[5];
  const float* b2      = (const float*)d_in[6];
  const float* Wr      = (const float*)d_in[7];
  const float* br      = (const float*)d_in[8];
  float* out = (float*)d_out;

  const int D = DIM, H = DIM / 2;
  const int B = in_sizes[0] / D;   // 16384

  // workspace carve-out (~180.5 MB total)
  char* ws = (char*)d_ws;
  size_t o = 0;
  auto carve = [&](size_t bytes) {
    void* p = ws + o;
    o = (o + bytes + 255) & ~(size_t)255;
    return p;
  };
  float*  sc  = (float*) carve((size_t)3 * B * 4);
  __bf16* Xb  = (__bf16*)carve((size_t)B * D * 2);
  __bf16* W1t = (__bf16*)carve((size_t)D * D * 2);
  __bf16* W2t = (__bf16*)carve((size_t)H * D * 2);
  __bf16* Hb  = (__bf16*)carve((size_t)B * D * 2);
  __bf16* Sb  = (__bf16*)carve((size_t)B * H * 2);

  // 1) per-row align/div/tens
  row_scalars_kernel<<<B, 256, 0, stream>>>(v_claim, v_doc, sc, B);
  // 2) h_final -> bf16
  cvt_bf16_kernel<<<8192, 256, 0, stream>>>(h_final, Xb, (long)B * D / 4);
  // 3) weights -> bf16 transposed [N][K]
  transpose_cvt_kernel<<<4096, 256, 0, stream>>>(W1, W1t, D, D);
  transpose_cvt_kernel<<<2048, 256, 0, stream>>>(W2, W2t, D, H);
  // 4) GEMM1: Hb = relu(Xb @ W1[0:2048] + b1 + align*W1[2048] + div*W1[2049] + tens*W1[2050])
  gemm_bf16_wmma<<<dim3(D / 128, B / 256), 256, 0, stream>>>(
      Xb, W1t, Hb, B, D, D, b1, sc, B, W1 + (size_t)D * D);
  // 5) GEMM2: Sb = relu(Hb @ W2 + b2)
  gemm_bf16_wmma<<<dim3(H / 128, B / 256), 256, 0, stream>>>(
      Hb, W2t, Sb, B, H, D, b2, nullptr, 0, nullptr);
  // 6) GEMV + sigmoid
  gemv_sigmoid_kernel<<<(B + 7) / 8, 256, 0, stream>>>(Sb, Wr, br, out, B, H);
}
